// SpikingSelfAttention_5806795784193
// MI455X (gfx1250) — matrix operand order repaired and checked by reading, not compiled, not run on hardware
//
#include <hip/hip_runtime.h>
#include <hip/hip_bf16.h>

// ---------------------------------------------------------------------------
// Spiking self-attention (SpikFormer SSA) for MI455X / gfx1250.
// Heavy GEMMs (binary-spike activations x f16 weights) run on
// v_wmma_f32_16x16x32_f16. Both GEMM tiles are staged with
// global_load_async_to_lds_b128 (ASYNCcnt path): the LIF kernels emit the
// spike tensor in transposed [b][n][c] layout so the B operand is
// contiguous along K and needs no LDS transpose. LIF scans are cheap
// bandwidth-bound elementwise passes.
// ---------------------------------------------------------------------------

typedef _Float16 half_t;
typedef __attribute__((ext_vector_type(16))) _Float16     v16h;
typedef __attribute__((ext_vector_type(8)))  float        v8f;
typedef __attribute__((ext_vector_type(4)))  unsigned int u32x4;

union Pack16 { u32x4 q[2]; half_t h[16]; v16h v; };
union Pack32 { u32x4 q[4]; half_t h[32]; };

static __device__ __forceinline__ v8f wmma_f16(v16h a, v16h b, v8f c) {
  return __builtin_amdgcn_wmma_f32_16x16x32_f16(false, a, false, b,
                                                (short)0, c, false, false);
}

// Async copy: 16 bytes global -> LDS, tracked on ASYNCcnt (no VGPR roundtrip).
static __device__ __forceinline__ void async_b128_to_lds(const void* gptr,
                                                         void* lds_ptr) {
  unsigned lds_off = (unsigned)(__SIZE_TYPE__)lds_ptr;
  asm volatile("global_load_async_to_lds_b128 %0, %1, off"
               :
               : "v"(lds_off), "v"(gptr)
               : "memory");
}
static __device__ __forceinline__ void wait_async0() {
  asm volatile("s_wait_asynccnt 0" ::: "memory");
}

// Problem constants
#define TT  4
#define BB  8
#define CC  512
#define NN  1024
#define HD  8      // heads
#define DD  64     // head dim
#define CHUNK ((size_t)BB * CC * NN)   // elements per timestep

// ---------------------------------------------------------------------------
// fp32 -> fp16 weight conversion
// ---------------------------------------------------------------------------
__global__ __launch_bounds__(256) void f32_to_f16_kernel(
    const float* __restrict__ a, half_t* __restrict__ o, int n) {
  int i = blockIdx.x * 256 + threadIdx.x;
  if (i < n) o[i] = (half_t)a[i];
}

// ---------------------------------------------------------------------------
// BN affine precompute: inv = gamma/sqrt(var+eps), shift = beta - mean*inv
// (+ bias*inv for the proj branch)
// ---------------------------------------------------------------------------
__global__ __launch_bounds__(256) void bn_params_kernel(
    const float* __restrict__ gamma, const float* __restrict__ beta,
    const float* __restrict__ mean,  const float* __restrict__ var,
    const float* __restrict__ bias,  // may be nullptr
    float* __restrict__ inv, float* __restrict__ shift) {
  int c = blockIdx.x * 256 + threadIdx.x;
  if (c < CC) {
    float iv = gamma[c] / sqrtf(var[c] + 1e-5f);
    float sh = beta[c] - mean[c] * iv;
    if (bias) sh += bias[c] * iv;
    inv[c] = iv;
    shift[c] = sh;
  }
}

// ---------------------------------------------------------------------------
// Multistep LIF: v' = 0.5*(v + x); s = H(v' - 1); v <- v' * (1 - s)
// in:  fp32 [T][b][c][n]
// out: s  (optional) f16 spikes, channel-major  [T][b][c][n]
//      st (optional) f16 spikes, token-major    [T][b][n][c]  (GEMM B layout)
// ---------------------------------------------------------------------------
__global__ __launch_bounds__(256) void lif_kernel(
    const float* __restrict__ x, half_t* __restrict__ s,
    half_t* __restrict__ st) {
  size_t i = (size_t)blockIdx.x * 256 + threadIdx.x;
  if (i >= CHUNK) return;
  const size_t b = i / ((size_t)CC * NN);
  const size_t c = (i / NN) % CC;
  const size_t n = i % NN;
  const size_t ti = (b * NN + n) * CC + c;  // transposed index within chunk
  float v = 0.f;
  #pragma unroll
  for (int t = 0; t < TT; ++t) {
    float xv = x[(size_t)t * CHUNK + i];
    v = 0.5f * (v + xv);
    float sp = (v >= 1.0f) ? 1.0f : 0.0f;
    if (s)  s[(size_t)t * CHUNK + i] = (half_t)sp;
    if (st) st[(size_t)t * CHUNK + ti] = (half_t)sp;
    v = (sp > 0.f) ? 0.f : v;
  }
}

// ---------------------------------------------------------------------------
// Batched channel GEMM + BN affine epilogue.
//   Y[b][o][n] = (sum_c W[o][c] * St[b][n][c]) * inv[o] + shift[o]
// W:  [512,512] f16 row-major (contiguous along K=c)
// St: [batch][1024 n][512 c] f16 (token-major spikes; contiguous along K=c)
// Y:  fp32 [batch][512 o][1024 n]
// Block tile 128(M) x 128(N), K-step 32, 8 waves (each 64x32 -> 8 WMMA tiles).
// Both tiles staged via global_load_async_to_lds_b128; all fragment reads are
// contiguous ds_load_b128 with k = 16*hi + s, the same order for A and B.
// ---------------------------------------------------------------------------
__global__ __launch_bounds__(256) void gemm512_bn_kernel(
    const half_t* __restrict__ W, const half_t* __restrict__ St,
    float* __restrict__ Y, const float* __restrict__ inv,
    const float* __restrict__ shift) {
  const int obase = blockIdx.x * 128;
  const int nbase = blockIdx.y * 128;
  const half_t* Sb = St + (size_t)blockIdx.z * NN * CC;
  float* Yb = Y + (size_t)blockIdx.z * CC * NN;

  __shared__ half_t lds_a[128 * 32];   // [m][k]
  __shared__ half_t lds_bT[128 * 32];  // [n][k]

  const int tid = threadIdx.x;
  const int wave = tid >> 5, lane = tid & 31;
  const int lm = lane & 15, hi = lane >> 4;
  const int wm = (wave & 1) * 64;   // wave M offset in block tile
  const int wn = (wave >> 1) * 32;  // wave N offset in block tile

  // per-thread staging coordinates (constant across K loop)
  const int row = tid >> 1, seg = (tid & 1) * 16;  // 128 rows x 32 k, 16 h/thr

  v8f acc[4][2] = {};

  for (int kk = 0; kk < CC; kk += 32) {
    // --- async stage A: W[obase+row][kk+seg .. +15] (x2 chunks) ----------
    {
      const half_t* src = W + (size_t)(obase + row) * CC + kk + seg;
      half_t* dst = &lds_a[row * 32 + seg];
      async_b128_to_lds(src, dst);
      async_b128_to_lds(src + 8, dst + 8);
    }
    // --- async stage B: St[nbase+row][kk+seg .. +15] (x2 chunks) ---------
    {
      const half_t* src = Sb + (size_t)(nbase + row) * CC + kk + seg;
      half_t* dst = &lds_bT[row * 32 + seg];
      async_b128_to_lds(src, dst);
      async_b128_to_lds(src + 8, dst + 8);
    }
    wait_async0();
    __syncthreads();

    // --- fragments + WMMA ------------------------------------------------
    Pack16 afr[4];
    #pragma unroll
    for (int i = 0; i < 4; ++i) {
      int m = wm + i * 16 + lm;
      afr[i].q[0] = *(const u32x4*)(&lds_a[m * 32 + hi * 16]);
      afr[i].q[1] = *(const u32x4*)(&lds_a[m * 32 + hi * 16 + 8]);
    }
    Pack16 bfr[2];
    #pragma unroll
    for (int j = 0; j < 2; ++j) {
      int n = wn + j * 16 + lm;
      const u32x4* p = (const u32x4*)(&lds_bT[n * 32 + hi * 16]);
      bfr[j].q[0] = p[0];
      bfr[j].q[1] = p[1];
    }
    #pragma unroll
    for (int i = 0; i < 4; ++i)
      #pragma unroll
      for (int j = 0; j < 2; ++j)
        acc[i][j] = wmma_f16(afr[i].v, bfr[j].v, acc[i][j]);
    __syncthreads();
  }

  // --- epilogue: BN affine, fp32 store ----------------------------------
  #pragma unroll
  for (int i = 0; i < 4; ++i) {
    #pragma unroll
    for (int j = 0; j < 2; ++j) {
      int n = nbase + wn + j * 16 + lm;
      #pragma unroll
      for (int r = 0; r < 8; ++r) {
        int o = obase + wm + i * 16 + hi * 8 + r;
        Yb[(size_t)o * NN + n] = acc[i][j][r] * inv[o] + shift[o];
      }
    }
  }
}

// ---------------------------------------------------------------------------
// kv[t,b,h][d][e] = sum_n k[c=h*64+d][n] * v[c=h*64+e][n]   (binary spikes)
// Both operands contiguous along K(=n): direct global b128 fragment loads.
// 1 block per (t,b,h); 4 waves, each one 16(d)-strip x 64(e).
// ---------------------------------------------------------------------------
__global__ __launch_bounds__(128) void kv_kernel(
    const half_t* __restrict__ Ks, const half_t* __restrict__ Vs,
    half_t* __restrict__ KV) {
  const int tbh = blockIdx.x;
  const int tb = tbh >> 3, h = tbh & 7;
  const half_t* kb = Ks + ((size_t)tb * CC + h * DD) * NN;
  const half_t* vb = Vs + ((size_t)tb * CC + h * DD) * NN;

  const int tid = threadIdx.x;
  const int wave = tid >> 5, lane = tid & 31;
  const int lm = lane & 15, hi = lane >> 4;
  const int m0 = wave * 16;  // d strip

  v8f acc[4] = {};
  for (int nk = 0; nk < NN; nk += 32) {
    Pack16 a;
    const half_t* ar = kb + (size_t)(m0 + lm) * NN + nk;
    a.q[0] = *(const u32x4*)(ar + hi * 16);
    a.q[1] = *(const u32x4*)(ar + hi * 16 + 8);
    #pragma unroll
    for (int j = 0; j < 4; ++j) {
      Pack16 b;
      const u32x4* p = (const u32x4*)(vb + (size_t)(j * 16 + lm) * NN + nk + hi * 16);
      b.q[0] = p[0];
      b.q[1] = p[1];
      acc[j] = wmma_f16(a.v, b.v, acc[j]);
    }
  }
  half_t* kvb = KV + (size_t)tbh * DD * DD;
  #pragma unroll
  for (int j = 0; j < 4; ++j)
    #pragma unroll
    for (int r = 0; r < 8; ++r)
      kvb[(m0 + hi * 8 + r) * DD + j * 16 + lm] = (half_t)acc[j][r];
}

// ---------------------------------------------------------------------------
// attT[c=h*64+e][n] = 0.125 * sum_d kv[d][e] * q[c=h*64+d][n]
// (== (q @ kv)^T, directly in the channel-major layout the next LIF needs)
// grid = (N/128, T*B*H); 4 waves, each one 16(e)-strip x 128(n).
// ---------------------------------------------------------------------------
__global__ __launch_bounds__(128) void att_kernel(
    const half_t* __restrict__ Qs, const half_t* __restrict__ KV,
    float* __restrict__ Att) {
  const int nbase = blockIdx.x * 128;
  const int tbh = blockIdx.y;
  const int tb = tbh >> 3, h = tbh & 7;

  __shared__ half_t kvT[DD * DD];   // [e][d]
  __shared__ half_t qT[128 * DD];   // [n][d]

  const int tid = threadIdx.x;

  // stage kv -> kvT (transpose)
  {
    const half_t* src = KV + (size_t)tbh * DD * DD;
    int d = tid >> 1, e0 = (tid & 1) * 32;
    Pack32 p;
    const u32x4* s4 = (const u32x4*)(src + d * DD + e0);
    p.q[0] = s4[0]; p.q[1] = s4[1]; p.q[2] = s4[2]; p.q[3] = s4[3];
    #pragma unroll
    for (int j = 0; j < 32; ++j) kvT[(e0 + j) * DD + d] = p.h[j];
  }
  // stage q tile -> qT (transpose): q rows are contiguous in n
  {
    const half_t* qb = Qs + ((size_t)tb * CC + h * DD) * NN;
    int d = tid >> 1, c0 = (tid & 1) * 64;
    #pragma unroll
    for (int seg = 0; seg < 8; ++seg) {
      Pack16 p;  // use first 8 halves only
      p.q[0] = *(const u32x4*)(qb + (size_t)d * NN + nbase + c0 + seg * 8);
      #pragma unroll
      for (int j = 0; j < 8; ++j) qT[(c0 + seg * 8 + j) * DD + d] = p.h[j];
    }
  }
  __syncthreads();

  const int wave = tid >> 5, lane = tid & 31;
  const int lm = lane & 15, hi = lane >> 4;
  const int m0 = wave * 16;  // e strip

  v8f acc[8] = {};
  #pragma unroll
  for (int k = 0; k < DD; k += 32) {
    Pack16 a;
    a.q[0] = *(const u32x4*)(&kvT[(m0 + lm) * DD + k + hi * 16]);
    a.q[1] = *(const u32x4*)(&kvT[(m0 + lm) * DD + k + hi * 16 + 8]);
    #pragma unroll
    for (int j = 0; j < 8; ++j) {
      Pack16 b;
      const u32x4* p = (const u32x4*)(&qT[(j * 16 + lm) * DD + k + hi * 16]);
      b.q[0] = p[0];
      b.q[1] = p[1];
      acc[j] = wmma_f16(a.v, b.v, acc[j]);
    }
  }

  float* ab = Att + ((size_t)tb * CC + h * DD) * NN;
  #pragma unroll
  for (int j = 0; j < 8; ++j)
    #pragma unroll
    for (int r = 0; r < 8; ++r)
      ab[(size_t)(m0 + hi * 8 + r) * NN + nbase + j * 16 + lm] =
          acc[j][r] * 0.125f;
}

// ---------------------------------------------------------------------------
// Host-side launcher
// ---------------------------------------------------------------------------
extern "C" void kernel_launch(void* const* d_in, const int* in_sizes, int n_in,
                              void* d_out, int out_size, void* d_ws,
                              size_t ws_size, hipStream_t stream) {
  (void)in_sizes; (void)n_in; (void)out_size; (void)ws_size;

  const float* x = (const float*)d_in[0];
  const float* w_f32[4]   = {(const float*)d_in[1],  (const float*)d_in[6],
                             (const float*)d_in[11], (const float*)d_in[16]};
  const float* gamma_p[4] = {(const float*)d_in[2],  (const float*)d_in[7],
                             (const float*)d_in[12], (const float*)d_in[17]};
  const float* beta_p[4]  = {(const float*)d_in[3],  (const float*)d_in[8],
                             (const float*)d_in[13], (const float*)d_in[18]};
  const float* mean_p[4]  = {(const float*)d_in[4],  (const float*)d_in[9],
                             (const float*)d_in[14], (const float*)d_in[19]};
  const float* var_p[4]   = {(const float*)d_in[5],  (const float*)d_in[10],
                             (const float*)d_in[15], (const float*)d_in[20]};
  const float* proj_b = (const float*)d_in[21];

  // workspace layout
  const size_t SPIKE_BYTES = (size_t)TT * CHUNK * sizeof(half_t);  // 33.5 MB
  const size_t W_BYTES = (size_t)CC * CC * sizeof(half_t);         // 0.5 MB
  const size_t KV_BYTES = (size_t)TT * BB * HD * DD * DD * sizeof(half_t);

  char* ws = (char*)d_ws;
  size_t off = 0;
  half_t* sT16 = (half_t*)(ws + off); off += SPIKE_BYTES;  // token-major s
  half_t* aT16 = (half_t*)(ws + off); off += SPIKE_BYTES;  // token-major a
  half_t* q16  = (half_t*)(ws + off); off += SPIKE_BYTES;  // channel-major
  half_t* k16  = (half_t*)(ws + off); off += SPIKE_BYTES;
  half_t* v16  = (half_t*)(ws + off); off += SPIKE_BYTES;
  half_t* w16[4];
  for (int i = 0; i < 4; ++i) { w16[i] = (half_t*)(ws + off); off += W_BYTES; }
  half_t* kv16 = (half_t*)(ws + off); off += KV_BYTES;
  float* invp  = (float*)(ws + off); off += 4 * CC * sizeof(float);
  float* shfp  = (float*)(ws + off); off += 4 * CC * sizeof(float);
  float* Yf    = (float*)(ws + off); off += (size_t)TT * CHUNK * sizeof(float);

  // 1. weight conversion + BN params
  const int wn = CC * CC;
  for (int i = 0; i < 4; ++i) {
    f32_to_f16_kernel<<<(wn + 255) / 256, 256, 0, stream>>>(w_f32[i], w16[i], wn);
    bn_params_kernel<<<2, 256, 0, stream>>>(
        gamma_p[i], beta_p[i], mean_p[i], var_p[i],
        (i == 3) ? proj_b : (const float*)nullptr, invp + i * CC, shfp + i * CC);
  }

  const int lif_blocks = (int)((CHUNK + 255) / 256);
  dim3 ggrid(CC / 128, NN / 128, TT * BB);   // (4, 8, 32)

  // 2. input LIF -> token-major spikes (GEMM B layout)
  lif_kernel<<<lif_blocks, 256, 0, stream>>>(x, (half_t*)nullptr, sT16);

  // 3. q / k / v branches: GEMM+BN then LIF (channel-major spikes out)
  half_t* spikes[3] = {q16, k16, v16};
  for (int br = 0; br < 3; ++br) {
    gemm512_bn_kernel<<<ggrid, 256, 0, stream>>>(w16[br], sT16, Yf,
                                                 invp + br * CC, shfp + br * CC);
    lif_kernel<<<lif_blocks, 256, 0, stream>>>(Yf, spikes[br],
                                               (half_t*)nullptr);
  }

  // 4. kv = k^T v  per (t,b,h)
  kv_kernel<<<TT * BB * HD, 128, 0, stream>>>(k16, v16, kv16);

  // 5. att^T = kv^T q * scale, then LIF -> token-major spikes
  att_kernel<<<dim3(NN / 128, TT * BB * HD), 128, 0, stream>>>(q16, kv16, Yf);
  lif_kernel<<<lif_blocks, 256, 0, stream>>>(Yf, (half_t*)nullptr, aT16);

  // 6. proj GEMM + bias + BN -> fp32 output
  gemm512_bn_kernel<<<ggrid, 256, 0, stream>>>(w16[3], aT16, (float*)d_out,
                                               invp + 3 * CC, shfp + 3 * CC);
}
